// GraphLaplacianAttention_36120674960002
// MI455X (gfx1250) — matrix operand order repaired and checked by reading, not compiled, not run on hardware
//
#include <hip/hip_runtime.h>
#include <hip/hip_bf16.h>

// ---------------------------------------------------------------------------
// GraphLaplacianAttention for MI455X (gfx1250, wave32, WMMA bf16 + TDM)
// B=1, N=2048, DIM=512, HEADS=8, HEAD_DIM=64
// ---------------------------------------------------------------------------

#define NTOK 2048
#define DM   512
#define NH   8
#define HD   64
#define MASKV (-3.402823466e38f)   // torch.finfo(float32).min

typedef __attribute__((ext_vector_type(16))) __bf16 v16bf;
typedef __attribute__((ext_vector_type(8)))  __bf16 v8bf;
typedef __attribute__((ext_vector_type(8)))  float  v8f;
typedef __attribute__((ext_vector_type(4))) unsigned int u32x4;
typedef __attribute__((ext_vector_type(8))) int i32x8;
typedef __attribute__((ext_vector_type(4))) int i32x4;

#if defined(__HIP_DEVICE_COMPILE__) && __has_builtin(__builtin_amdgcn_tensor_load_to_lds) && __has_builtin(__builtin_amdgcn_s_wait_tensorcnt)
#define HAVE_TDM 1
#else
#define HAVE_TDM 0
#endif

union V16 { v16bf v; v8bf h[2]; };

static __device__ inline v8f zero8() {
  v8f z = {0.f,0.f,0.f,0.f,0.f,0.f,0.f,0.f};
  return z;
}

// A-matrix fragment (16x32 bf16, M x K), per ISA 7.12.2:
// lane16 = M row; halves 0..7  -> K = kbase + 8*g + (0..7)
//                 halves 8..15 -> K = kbase + 16 + 8*g + (0..7)
static __device__ inline v16bf load_a_frag(const __bf16* __restrict__ base, int ld,
                                           int m0, int kbase, int lane16, int g) {
  V16 u;
  const __bf16* p = base + (size_t)(m0 + lane16) * ld + kbase + 8 * g;
  u.h[0] = *(const v8bf*)(p);
  u.h[1] = *(const v8bf*)(p + 16);
  return u.v;
}

// B-matrix fragment (32x16 bf16, K x N) where B[k][n] = W[n][k] (W row-major NxK):
// lane16 = N col; halves 0..7 -> K = kbase + 8*g + (0..7); 8..15 -> K = kbase + 16 + 8*g + (0..7)
static __device__ inline v16bf load_b_frag(const __bf16* __restrict__ base, int ld,
                                           int n0, int kbase, int lane16, int g) {
  V16 u;
  const __bf16* p = base + (size_t)(n0 + lane16) * ld + kbase + 8 * g;
  u.h[0] = *(const v8bf*)(p);
  u.h[1] = *(const v8bf*)(p + 16);
  return u.v;
}

static __device__ inline v8f wmma_bf16(v16bf a, v16bf b, v8f c) {
  return __builtin_amdgcn_wmma_f32_16x16x32_bf16(false, a, false, b, (short)0, c,
                                                 false, false);
}

// ---------------------------------------------------------------------------
// TDM: 2D tile load global->LDS.  Tile = 64 rows x 64 bf16, row stride src_ld
// (elements).  D# per CDNA5 ISA 8.3/8.4: group0 = {count, lds_addr,
// global_addr, type=2}; group1 = {data_size=2B, tensor/tile dims, stride}.
// Issued by one wave; completion tracked with TENSORcnt.
// ---------------------------------------------------------------------------
#if HAVE_TDM
static __device__ inline void tdm_load_tile64(__bf16* lds_dst, const __bf16* gsrc,
                                              unsigned src_ld_elems) {
  const unsigned lds_off = (unsigned)(unsigned long long)(uintptr_t)lds_dst;
  const unsigned long long ga = (unsigned long long)(uintptr_t)gsrc;
  u32x4 g0 = {0u, 0u, 0u, 0u};
  g0.x = 1u;                                              // count=1, user mode
  g0.y = lds_off;                                         // lds_addr [63:32]
  g0.z = (unsigned)(ga & 0xFFFFFFFFull);                  // global_addr lo
  g0.w = (unsigned)((ga >> 32) & 0x01FFFFFFull)           // global_addr hi (57b)
       | 0x80000000u;                                     // type=2 ("image")
  i32x8 g1;
  g1[0] = (int)(1u << 16);                                // data_size=1 -> 2 bytes
  g1[1] = (int)((64u & 0xFFFFu) << 16);                   // tensor_dim0[15:0]=64
  g1[2] = (int)((64u >> 16) | ((64u & 0xFFFFu) << 16));   // dim0 hi | tensor_dim1 lo
  g1[3] = (int)((64u >> 16) | (64u << 16));               // dim1 hi | tile_dim0=64
  g1[4] = (int)(64u);                                     // tile_dim1=64, tile_dim2=0
  g1[5] = (int)src_ld_elems;                              // tensor_dim0_stride lo
  g1[6] = 0;                                              // stride0 hi | stride1 lo
  g1[7] = 0;
  i32x4 gz = {0, 0, 0, 0};
#if defined(__clang_major__) && (__clang_major__ >= 23)
  i32x8 gz8 = {0, 0, 0, 0, 0, 0, 0, 0};
  __builtin_amdgcn_tensor_load_to_lds(g0, g1, gz, gz, gz8, 0);
#else
  __builtin_amdgcn_tensor_load_to_lds(g0, g1, gz, gz, 0);
#endif
}
#endif

// Fallback cooperative copy (256 threads): 64x64 bf16 tile.
static __device__ inline void coop_copy_tile64(__bf16* dst, const __bf16* src,
                                               int src_ld, int tid) {
  const int row = tid >> 2;
  const int c0 = (tid & 3) * 16;
  *(v8bf*)(dst + row * 64 + c0)     = *(const v8bf*)(src + (size_t)row * src_ld + c0);
  *(v8bf*)(dst + row * 64 + c0 + 8) = *(const v8bf*)(src + (size_t)row * src_ld + c0 + 8);
}

// ---------------------------------------------------------------------------
// fp32 -> bf16 conversion
// ---------------------------------------------------------------------------
__global__ void cvt_bf16_kernel(const float* __restrict__ s, __bf16* __restrict__ d, int n) {
  int i = blockIdx.x * 256 + threadIdx.x;
  if (i < n) d[i] = (__bf16)s[i];
}

// ---------------------------------------------------------------------------
// Dual-source GEMM:  C = alpha * (A1 @ W1^T [+ A2 @ W2^T]) + bias
// Fully unrolled, register-pipelined k-loop: loads for k+32 are issued before
// the k WMMAs; unrolling lets SSA renaming absorb the stage rotation (no
// v_mov chains).  Template flags remove all inner uniform branches.
// Block = 256 thr = 8 waves; wave tile = 64(M) x 16(N); grid = (N/128, 2048/64).
// ---------------------------------------------------------------------------
template <bool DUAL, bool HASF, bool HASB, bool HASBT>
__global__ __launch_bounds__(256)
void gemm_kernel(const __bf16* __restrict__ A1, const __bf16* __restrict__ W1,
                 const __bf16* __restrict__ A2, const __bf16* __restrict__ W2,
                 const float* __restrict__ bias, float alpha,
                 float* __restrict__ outF, __bf16* __restrict__ outB,
                 __bf16* __restrict__ outBT) {
  const int lane   = threadIdx.x & 31;
  const int wave   = threadIdx.x >> 5;
  const int lane16 = lane & 15;
  const int g      = lane >> 4;
  const int n0 = (blockIdx.x * 8 + wave) * 16;
  const int m0 = blockIdx.y * 64;

  v8f acc[4];
#pragma unroll
  for (int i = 0; i < 4; ++i) acc[i] = zero8();

  // --- prologue: fragments for k=0 ---
  v16bf b1 = load_b_frag(W1, DM, n0, 0, lane16, g);
  v16bf a1[4];
#pragma unroll
  for (int i = 0; i < 4; ++i) a1[i] = load_a_frag(A1, DM, m0 + 16 * i, 0, lane16, g);
  v16bf b2{};
  v16bf a2[4];
  if constexpr (DUAL) {
    b2 = load_b_frag(W2, DM, n0, 0, lane16, g);
#pragma unroll
    for (int i = 0; i < 4; ++i) a2[i] = load_a_frag(A2, DM, m0 + 16 * i, 0, lane16, g);
  }

#pragma unroll
  for (int kb = 0; kb < DM; kb += 32) {
    const int kn = kb + 32;
    v16bf b1n{}, b2n{};
    v16bf a1n[4], a2n[4];
    if (kn < DM) {  // compile-time after unroll: prefetch next-step fragments
      b1n = load_b_frag(W1, DM, n0, kn, lane16, g);
#pragma unroll
      for (int i = 0; i < 4; ++i) a1n[i] = load_a_frag(A1, DM, m0 + 16 * i, kn, lane16, g);
      if constexpr (DUAL) {
        b2n = load_b_frag(W2, DM, n0, kn, lane16, g);
#pragma unroll
        for (int i = 0; i < 4; ++i) a2n[i] = load_a_frag(A2, DM, m0 + 16 * i, kn, lane16, g);
      }
    }
#pragma unroll
    for (int i = 0; i < 4; ++i) acc[i] = wmma_bf16(a1[i], b1, acc[i]);
    if constexpr (DUAL) {
#pragma unroll
      for (int i = 0; i < 4; ++i) acc[i] = wmma_bf16(a2[i], b2, acc[i]);
    }
    if (kn < DM) {
      b1 = b1n;
      b2 = b2n;
#pragma unroll
      for (int i = 0; i < 4; ++i) { a1[i] = a1n[i]; a2[i] = a2n[i]; }
    }
  }

  const float bb = bias ? bias[n0 + lane16] : 0.f;
  const int n = n0 + lane16;
#pragma unroll
  for (int i = 0; i < 4; ++i) {
#pragma unroll
    for (int r = 0; r < 8; ++r) {
      float v = acc[i][r] * alpha + bb;
      acc[i][r] = v;
      const int m = m0 + 16 * i + r + 8 * g;   // C layout: M = r + 8*(lane/16)
      if constexpr (HASF) outF[(size_t)m * DM + n] = v;
      if constexpr (HASB) outB[(size_t)m * DM + n] = (__bf16)v;
    }
    if constexpr (HASBT) {
      v8bf pk;
#pragma unroll
      for (int r = 0; r < 8; ++r) pk[r] = (__bf16)acc[i][r];
      *(v8bf*)(outBT + (size_t)n * NTOK + m0 + 16 * i + 8 * g) = pk;  // 16B aligned
    }
  }
}

// ---------------------------------------------------------------------------
// Fused graph-Laplacian flash attention with TDM-staged K/V tiles.
//   S = (Q @ KE^T) * conv_w[h];  S = mask ? MASK_VALUE : S
//   P = online_softmax(S);  out = VE - P @ VE   (Laplacian I - A folded in)
// Block = 256 thr = 8 waves, each wave = one 16-query stripe of one head.
// Per key block (64 keys): wave 0 issues tensor_load_to_lds for the NEXT
// block's KE(64x64) and VE(64x64) tiles into the ping-pong buffer while all
// waves run WMMAs on the current one; s_wait_tensorcnt + barrier publishes it.
// Grid = (2048/128, 8 heads).
// ---------------------------------------------------------------------------
__global__ __launch_bounds__(256)
void attn_kernel(const __bf16* __restrict__ Qb,    // [2048][512] bf16
                 const __bf16* __restrict__ KEb,   // [2048][512] bf16 (pre-scaled)
                 const __bf16* __restrict__ VEbT,  // [512][2048] bf16 transposed
                 const float*  __restrict__ VEf,   // [2048][512] f32
                 const unsigned char* __restrict__ mask,  // [2048*2048] bool
                 const float* __restrict__ conv_w,        // [8]
                 __bf16* __restrict__ AOb)                // [2048][512] bf16
{
  __shared__ __align__(16) __bf16 keTile[2][64 * 64];  // [key][dim], 8KB each
  __shared__ __align__(16) __bf16 veTile[2][64 * 64];  // [dim][key], 8KB each
  __shared__ __align__(16) __bf16 ldsP[8][16 * 72];    // per-wave P transpose

  const int tid    = threadIdx.x;
  const int lane   = tid & 31;
  const int wave   = tid >> 5;
  const int lane16 = lane & 15;
  const int g      = lane >> 4;
  const int q0 = blockIdx.x * 128 + wave * 16;
  const int h  = blockIdx.y;
  const int hd = h * HD;
  const float cw = conv_w[h];
  const int NB = NTOK / 64;

  // Q A-fragments for the two K=32 steps of the 64-wide head dim (regs)
  const v16bf qa0 = load_a_frag(Qb, DM, q0, hd + 0,  lane16, g);
  const v16bf qa1 = load_a_frag(Qb, DM, q0, hd + 32, lane16, g);

  v8f acc[4];
#pragma unroll
  for (int i = 0; i < 4; ++i) acc[i] = zero8();
  float m8[8], l8[8];
#pragma unroll
  for (int r = 0; r < 8; ++r) { m8[r] = -INFINITY; l8[r] = 0.f; }

  __bf16* pt = &ldsP[wave][0];

  // ---- stage key block 0 into buffer 0 ----
#if HAVE_TDM
  if (wave == 0) {
    tdm_load_tile64(&keTile[0][0], KEb + (size_t)0 * DM + hd, DM);
    tdm_load_tile64(&veTile[0][0], VEbT + (size_t)hd * NTOK + 0, NTOK);
    __builtin_amdgcn_s_wait_tensorcnt(0);
  }
#else
  coop_copy_tile64(&keTile[0][0], KEb + hd, DM, tid);
  coop_copy_tile64(&veTile[0][0], VEbT + (size_t)hd * NTOK, NTOK, tid);
#endif
  __syncthreads();

  for (int kb = 0; kb < NB; ++kb) {
    const int key0 = kb * 64;
    const int cur = kb & 1;
    const int nxt = cur ^ 1;
    const bool have_next = (kb + 1 < NB);
    const __bf16* ke = &keTile[cur][0];
    const __bf16* ve = &veTile[cur][0];

#if HAVE_TDM
    // async DMA of next tiles overlaps with this block's WMMAs
    if (wave == 0 && have_next) {
      tdm_load_tile64(&keTile[nxt][0], KEb + (size_t)(key0 + 64) * DM + hd, DM);
      tdm_load_tile64(&veTile[nxt][0], VEbT + (size_t)hd * NTOK + key0 + 64, NTOK);
    }
#endif

    // ---- S = Q @ KE^T from LDS tile (4 fragments of 16x16) ----
    v8f S[4];
#pragma unroll
    for (int fc = 0; fc < 4; ++fc) {
      v8f c = zero8();
      c = wmma_bf16(qa0, load_b_frag(ke, 64, fc * 16, 0,  lane16, g), c);
      c = wmma_bf16(qa1, load_b_frag(ke, 64, fc * 16, 32, lane16, g), c);
      S[fc] = c;
    }

    // ---- per-head conv scale + mask (mask==true -> MASK_VALUE) ----
    const unsigned char* mrow = mask + (size_t)(q0 + 8 * g) * NTOK + key0;
#pragma unroll
    for (int fc = 0; fc < 4; ++fc) {
#pragma unroll
      for (int r = 0; r < 8; ++r) {
        float s = S[fc][r] * cw;
        if (mrow[(size_t)r * NTOK + fc * 16 + lane16]) s = MASKV;
        S[fc][r] = s;
      }
    }

    // ---- online softmax (rows live across 16 lanes of each half-wave) ----
    float nm[8];
#pragma unroll
    for (int r = 0; r < 8; ++r) {
      float v = fmaxf(fmaxf(S[0][r], S[1][r]), fmaxf(S[2][r], S[3][r]));
      v = fmaxf(v, __shfl_xor(v, 1));
      v = fmaxf(v, __shfl_xor(v, 2));
      v = fmaxf(v, __shfl_xor(v, 4));
      v = fmaxf(v, __shfl_xor(v, 8));
      nm[r] = fmaxf(m8[r], v);
    }
#pragma unroll
    for (int r = 0; r < 8; ++r) {
      float rs = 0.f;
#pragma unroll
      for (int fc = 0; fc < 4; ++fc) {
        float p = __expf(S[fc][r] - nm[r]);
        S[fc][r] = p;
        rs += p;
      }
      rs += __shfl_xor(rs, 1);
      rs += __shfl_xor(rs, 2);
      rs += __shfl_xor(rs, 4);
      rs += __shfl_xor(rs, 8);
      const float alpha = __expf(m8[r] - nm[r]);
      l8[r] = l8[r] * alpha + rs;
      m8[r] = nm[r];
#pragma unroll
      for (int fd = 0; fd < 4; ++fd) acc[fd][r] *= alpha;
    }

    // ---- transpose P (C layout) -> A layout via per-wave LDS tile ----
#pragma unroll
    for (int fc = 0; fc < 4; ++fc)
#pragma unroll
      for (int r = 0; r < 8; ++r)
        pt[(r + 8 * g) * 72 + fc * 16 + lane16] = (__bf16)S[fc][r];
    asm volatile("s_wait_dscnt 0" ::: "memory");  // cross-lane LDS RAW, same wave

    V16 pa0, pa1;
    {
      const __bf16* pr = pt + lane16 * 72;  // M = lane16, K = key-in-block
      pa0.h[0] = *(const v8bf*)(pr + 8 * g);
      pa0.h[1] = *(const v8bf*)(pr + 16 + 8 * g);
      pa1.h[0] = *(const v8bf*)(pr + 32 + 8 * g);
      pa1.h[1] = *(const v8bf*)(pr + 48 + 8 * g);
    }

    // ---- acc += P @ VE_block from LDS tile ----
#pragma unroll
    for (int fd = 0; fd < 4; ++fd) {
      acc[fd] = wmma_bf16(pa0.v, load_b_frag(ve, 64, fd * 16, 0,  lane16, g), acc[fd]);
      acc[fd] = wmma_bf16(pa1.v, load_b_frag(ve, 64, fd * 16, 32, lane16, g), acc[fd]);
    }

#if HAVE_TDM
    if (wave == 0 && have_next) __builtin_amdgcn_s_wait_tensorcnt(0);
    __syncthreads();   // publish next tile / protect buffers
#else
    __syncthreads();
    if (have_next) {
      coop_copy_tile64(&keTile[nxt][0], KEb + (size_t)(key0 + 64) * DM + hd, DM, tid);
      coop_copy_tile64(&veTile[nxt][0], VEbT + (size_t)hd * NTOK + key0 + 64, NTOK, tid);
    }
    __syncthreads();
#endif
  }

  // ---- epilogue: out = VE - P@VE  (graph Laplacian (I - A) @ VE) ----
#pragma unroll
  for (int fd = 0; fd < 4; ++fd) {
#pragma unroll
    for (int r = 0; r < 8; ++r) {
      const int q = q0 + r + 8 * g;
      const int d = hd + fd * 16 + lane16;
      const float v = VEf[(size_t)q * DM + d] - acc[fd][r] / l8[r];
      AOb[(size_t)q * DM + d] = (__bf16)v;
    }
  }
}

// ---------------------------------------------------------------------------
// Host launcher
// ---------------------------------------------------------------------------
extern "C" void kernel_launch(void* const* d_in, const int* in_sizes, int n_in,
                              void* d_out, int out_size, void* d_ws, size_t ws_size,
                              hipStream_t stream) {
  const float* x      = (const float*)d_in[0];   // [1,2048,512]
  const float* edges  = (const float*)d_in[1];   // [1,2048,512]
  const unsigned char* mask = (const unsigned char*)d_in[2];  // [1,1,2048,2048] bool
  const float* Wqk    = (const float*)d_in[3];   // [1024,512]
  const float* Wv     = (const float*)d_in[4];   // [512,512]
  const float* bv     = (const float*)d_in[5];   // [512]
  const float* Wek    = (const float*)d_in[6];   // [512,512]
  const float* Wev    = (const float*)d_in[7];   // [512,512]
  const float* conv_w = (const float*)d_in[8];   // [8]
  const float* Wo     = (const float*)d_in[9];   // [512,512]
  const float* bo     = (const float*)d_in[10];  // [512]
  float* out = (float*)d_out;                    // [1,2048,512] f32

  // workspace carve (~19.5 MB)
  char* p = (char*)d_ws;
  auto carve = [&](size_t bytes) -> char* {
    char* r = p;
    p += (bytes + 255) & ~(size_t)255;
    return r;
  };
  __bf16* xb   = (__bf16*)carve((size_t)NTOK * DM * 2);
  __bf16* eb   = (__bf16*)carve((size_t)NTOK * DM * 2);
  __bf16* Wqkb = (__bf16*)carve((size_t)2 * DM * DM * 2);
  __bf16* Wvb  = (__bf16*)carve((size_t)DM * DM * 2);
  __bf16* Wekb = (__bf16*)carve((size_t)DM * DM * 2);
  __bf16* Wevb = (__bf16*)carve((size_t)DM * DM * 2);
  __bf16* Wob  = (__bf16*)carve((size_t)DM * DM * 2);
  __bf16* Qb   = (__bf16*)carve((size_t)NTOK * DM * 2);
  __bf16* KEb  = (__bf16*)carve((size_t)NTOK * DM * 2);
  __bf16* VEbT = (__bf16*)carve((size_t)DM * NTOK * 2);
  float*  VEf  = (float*) carve((size_t)NTOK * DM * 4);
  __bf16* AOb  = (__bf16*)carve((size_t)NTOK * DM * 2);

  auto cvt = [&](const float* s, __bf16* d, int n) {
    cvt_bf16_kernel<<<(n + 255) / 256, 256, 0, stream>>>(s, d, n);
  };
  cvt(x,     xb,   NTOK * DM);
  cvt(edges, eb,   NTOK * DM);
  cvt(Wqk,   Wqkb, 2 * DM * DM);
  cvt(Wv,    Wvb,  DM * DM);
  cvt(Wek,   Wekb, DM * DM);
  cvt(Wev,   Wevb, DM * DM);
  cvt(Wo,    Wob,  DM * DM);

  const dim3 ggrid(DM / 128, NTOK / 64);
  const float scale = 0.125f;  // HEAD_DIM^-0.5, folded into KE (scales k and edge_k)

  // Q = x @ Wqk[0:512].T                          (bf16 row-major out)
  gemm_kernel<false, false, true, false><<<ggrid, 256, 0, stream>>>(
      xb, Wqkb, nullptr, nullptr, nullptr, 1.0f, nullptr, Qb, nullptr);
  // KE = scale * (x @ Wqk[512:1024].T + edges @ Wek.T)   (bf16 row-major out)
  gemm_kernel<true, false, true, false><<<ggrid, 256, 0, stream>>>(
      xb, Wqkb + (size_t)DM * DM, eb, Wekb, nullptr, scale, nullptr, KEb, nullptr);
  // VE = x @ Wv.T + bv + edges @ Wev.T    (f32 row-major + bf16 transposed)
  gemm_kernel<true, true, false, true><<<ggrid, 256, 0, stream>>>(
      xb, Wvb, eb, Wevb, bv, 1.0f, VEf, nullptr, VEbT);

  // fused Laplacian flash attention -> AOb (bf16 [2048][512])
  attn_kernel<<<dim3(NTOK / 128, NH), 256, 0, stream>>>(Qb, KEb, VEbT, VEf,
                                                        mask, conv_w, AOb);

  // out = AO @ Wo.T + bo   (f32 final output)
  gemm_kernel<false, true, false, false><<<ggrid, 256, 0, stream>>>(
      AOb, Wob, nullptr, nullptr, bo, 1.0f, out, nullptr, nullptr);
}
